// DCGRU_4183298146748
// MI455X (gfx1250) — compile-verified, matrix operand order checked by reading, not run on hardware
//
#include <hip/hip_runtime.h>
#include <hip/hip_bf16.h>

// ---------------------------------------------------------------------------
// DCGRU (DCRNN) forward on gfx1250.
//   B=32, L=12, N=2000, C=2, H=64, E=32000, S=2, K=2
//   CAT = 66, N_MATS = 5, K-dim = 330 (padded to 352 for 32-wide WMMA chunks)
//
// Per timestep (7 launches):
//   concat -> spmm(x1) -> spmm(x2) -> gate GEMM (WMMA, fused sigmoid + rebuild
//   feat with v2=[r*h, x_t]) -> spmm(x1) -> spmm(x2) -> cand GEMM (WMMA, fused
//   tanh + GRU update, writes h and out[:,t]).
//
// WMMA data path:
//   * W_gate / W_cand pre-packed once per launch into fragment-ready f16
//     (per k-chunk / N-tile / lane order of the 16-bit B layout) -> B fragment
//     is two coalesced global_load_b128 per tile, no LDS, no packing ALU.
//   * feat block staged fp32->f16 into LDS in a K-permuted layout so each
//     lane's A fragment is 32 contiguous bytes -> 2x ds_load_b128.
// ---------------------------------------------------------------------------

typedef __attribute__((ext_vector_type(16))) _Float16 v16h;
typedef __attribute__((ext_vector_type(8)))  float    v8f;

#define DB   32
#define DL   12
#define DN   2000
#define DC   2
#define DH   64
#define DE   32000
#define DS   2
#define DCAT 66
#define DKD  330
#define DKP  352          // 11 chunks of 32
#define DKC  (DKP / 32)   // 11
#define DM   (DB * DN)    // 64000
#define NGATE 128
#define NCAND 64

union H8 { _Float16 h[8]; uint4 u; };

// ---------------------------------------------------------------------------
// init: copy h_0 -> h workspace, zero the K padding columns (330..351)
// ---------------------------------------------------------------------------
__global__ void dcgru_init_kernel(float* __restrict__ h,
                                  const float* __restrict__ h0,
                                  float* __restrict__ feat) {
    int idx = blockIdx.x * blockDim.x + threadIdx.x;
    const int nH = DM * DH;
    const int nP = DM * (DKP - DKD);
    if (idx < nH) {
        h[idx] = h0[idx];
    } else if (idx < nH + nP) {
        int j   = idx - nH;
        int row = j / (DKP - DKD);
        int c   = j - row * (DKP - DKD);
        feat[(size_t)row * DKP + DKD + c] = 0.0f;
    }
}

// ---------------------------------------------------------------------------
// Pre-pack W (KD x NOUT fp32) into fragment-ready f16:
//   Wf[((kc*NT + nt)*32 + lane)*16 + j]
//   j<8 : k = kc*32 + lh*8 + j        (lh = lane>>4, col = nt*16 + (lane&15))
//   j>=8: k = kc*32 + 16 + lh*8 + j-8
// ---------------------------------------------------------------------------
template <int NOUT>
__global__ void dcgru_pack_w_kernel(const float* __restrict__ W,
                                    _Float16* __restrict__ Wf) {
    const int NT = NOUT / 16;
    int idx = blockIdx.x * blockDim.x + threadIdx.x;
    if (idx >= DKC * NT * 32 * 16) return;
    int j    = idx & 15;
    int lane = (idx >> 4) & 31;
    int rest = idx >> 9;
    int nt   = rest % NT;
    int kc   = rest / NT;
    int lh   = lane >> 4;
    int l15  = lane & 15;
    int kin  = (j < 8) ? (lh * 8 + j) : (16 + lh * 8 + (j - 8));
    int k    = kc * 32 + kin;
    int col  = nt * 16 + l15;
    float v  = (k < DKD) ? W[(size_t)k * NOUT + col] : 0.0f;
    Wf[idx] = (_Float16)v;
}

// ---------------------------------------------------------------------------
// concat: v = [h, x_t]; feat row: [0:66]=v [66:132]=0 [132:198]=-v
//                                 [198:264]=0 [264:330]=-v
// ---------------------------------------------------------------------------
__global__ void dcgru_concat_kernel(float* __restrict__ feat,
                                    const float* __restrict__ h,
                                    const float* __restrict__ x,
                                    int t) {
    int idx = blockIdx.x * blockDim.x + threadIdx.x;
    if (idx >= DM * DCAT) return;
    int row = idx / DCAT;
    int f   = idx - row * DCAT;
    float v;
    if (f < DH) {
        v = h[(size_t)row * DH + f];
    } else {
        int b = row / DN;
        int node = row - b * DN;
        v = x[(((size_t)b * DL + t) * DN + node) * DC + (f - DH)];
    }
    float* fr = feat + (size_t)row * DKP;
    fr[f]       = v;
    fr[66 + f]  = 0.0f;
    fr[132 + f] = -v;
    fr[198 + f] = 0.0f;
    fr[264 + f] = -v;
}

// ---------------------------------------------------------------------------
// spmm: phase 1: x1 += w * v ; phase 2: x2 += 2w * x1 (x2 pre-seeded -x0)
// ---------------------------------------------------------------------------
__global__ void dcgru_spmm_kernel(float* __restrict__ feat,
                                  const int*   __restrict__ As_idx,
                                  const float* __restrict__ As_w,
                                  int phase) {
    int idx = blockIdx.x * blockDim.x + threadIdx.x;
    if (idx >= DE * DCAT) return;
    int e = idx / DCAT;
    int f = idx - e * DCAT;
    int s = blockIdx.y;

    int   src = As_idx[(size_t)s * 2 * DE + e];
    int   dst = As_idx[(size_t)s * 2 * DE + DE + e];
    float w   = As_w[(size_t)s * DE + e];

    int   sb, db;
    float scale;
    if (phase == 1) { sb = 0;            db = 66 + 132 * s;  scale = 1.0f; }
    else            { sb = 66 + 132 * s; db = 132 + 132 * s; scale = 2.0f; }
    float sw = scale * w;

    const size_t srcoff = (size_t)src * DKP + sb + f;
    const size_t dstoff = (size_t)dst * DKP + db + f;
    #pragma unroll 4
    for (int b = 0; b < DB; ++b) {
        size_t rb = (size_t)b * DN * DKP;
        float v = feat[rb + srcoff];
        atomicAdd(&feat[rb + dstoff], sw * v);
    }
}

// ---------------------------------------------------------------------------
// WMMA mainloop: acc[128 x NOUT] = A[128 x 352] * B[352 x NOUT]
// sA permuted so each lane's A fragment is 16 contiguous halves.
// B fragments read directly from pre-packed global Wf.
// ---------------------------------------------------------------------------
template <int NOUT>
__device__ __forceinline__ void gemm_block(const float* __restrict__ feat,
                                           const _Float16* __restrict__ Wf,
                                           int rowbase, v8f* acc,
                                           _Float16* sA) {
    const int tid  = threadIdx.x;
    const int lane = tid & 31;
    const int wave = tid >> 5;
    const int lh   = lane >> 4;
    const int l15  = lane & 15;
    const int NT   = NOUT / 16;

    v8f zero = {};
    #pragma unroll
    for (int nt = 0; nt < NT; ++nt) acc[nt] = zero;

    const int srow = tid >> 1;   // staging row (0..127)
    const int shf  = tid & 1;    // staging K half (0..1)
    const float* srcbase = feat + (size_t)(rowbase + srow) * DKP + shf * 16;

    for (int kc = 0; kc < DKC; ++kc) {
        // ---- stage A block: 128 rows x 32 k, fp32 -> f16, K-permuted ----
        {
            const float* srcp = srcbase + kc * 32;
            float4 p0 = ((const float4*)srcp)[0];
            float4 p1 = ((const float4*)srcp)[1];
            float4 p2 = ((const float4*)srcp)[2];
            float4 p3 = ((const float4*)srcp)[3];
            H8 g0, g1;
            g0.h[0]=(_Float16)p0.x; g0.h[1]=(_Float16)p0.y;
            g0.h[2]=(_Float16)p0.z; g0.h[3]=(_Float16)p0.w;
            g0.h[4]=(_Float16)p1.x; g0.h[5]=(_Float16)p1.y;
            g0.h[6]=(_Float16)p1.z; g0.h[7]=(_Float16)p1.w;
            g1.h[0]=(_Float16)p2.x; g1.h[1]=(_Float16)p2.y;
            g1.h[2]=(_Float16)p2.z; g1.h[3]=(_Float16)p2.w;
            g1.h[4]=(_Float16)p3.x; g1.h[5]=(_Float16)p3.y;
            g1.h[6]=(_Float16)p3.z; g1.h[7]=(_Float16)p3.w;
            // hf=0: k 0..7 -> +0,  k 8..15  -> +16
            // hf=1: k16..23 -> +8, k 24..31 -> +24
            *(uint4*)(sA + srow * 32 + shf * 8)      = g0.u;
            *(uint4*)(sA + srow * 32 + 16 + shf * 8) = g1.u;
            if (kc + 1 < DKC)                 // emits global_prefetch_b8
                __builtin_prefetch(srcp + 32, 0, 0);
        }
        __syncthreads();

        // ---- A fragment: 32 contiguous bytes from LDS (2x ds_load_b128) ----
        v16h a = *(const v16h*)(sA + (wave * 16 + l15) * 32 + lh * 16);

        // ---- B fragments straight from pre-packed global + WMMA ----
        const _Float16* wfk = Wf + (size_t)(kc * NT) * 512 + lane * 16;
        #pragma unroll
        for (int nt = 0; nt < NT; ++nt) {
            v16h b = *(const v16h*)(wfk + nt * 512);
            acc[nt] = __builtin_amdgcn_wmma_f32_16x16x32_f16(
                false, a, false, b, (short)0, acc[nt], false, false);
        }
        __syncthreads();
    }
}

// ---------------------------------------------------------------------------
// gate GEMM: gate = sigmoid(feat @ Wg + bg); writes u, rebuilds feat with v2
// ---------------------------------------------------------------------------
__global__ void __launch_bounds__(256)
dcgru_gemm_gate_kernel(float* __restrict__ feat,
                       const _Float16* __restrict__ Wf,
                       const float* __restrict__ bg,
                       const float* __restrict__ h,
                       const float* __restrict__ x,
                       float* __restrict__ u,
                       int t) {
    __shared__ __align__(32) _Float16 sA[128 * 32];
    v8f acc[NGATE / 16];
    const int rowbase = blockIdx.x * 128;

    gemm_block<NGATE>(feat, Wf, rowbase, acc, sA);

    const int lane = threadIdx.x & 31;
    const int wave = threadIdx.x >> 5;
    const int lh   = lane >> 4;
    const int l15  = lane & 15;
    const int R0   = rowbase + wave * 16;

    #pragma unroll
    for (int nt = 0; nt < NGATE / 16; ++nt) {
        int col = nt * 16 + l15;
        float bias = bg[col];
        #pragma unroll
        for (int r = 0; r < 8; ++r) {
            int R = R0 + r + 8 * lh;
            float sig = 1.0f / (1.0f + __expf(-(acc[nt][r] + bias)));
            float* fr = feat + (size_t)R * DKP;
            if (col < DH) {
                float v2 = sig * h[(size_t)R * DH + col];   // r_t * h_prev
                fr[col] = v2;  fr[66 + col] = 0.0f;
                fr[132 + col] = -v2; fr[198 + col] = 0.0f; fr[264 + col] = -v2;
            } else {
                u[(size_t)R * DH + (col - DH)] = sig;       // u_t
                if (col < DCAT) {                           // x_t channels
                    int b = R / DN, node = R - b * DN;
                    float xv = x[(((size_t)b * DL + t) * DN + node) * DC + (col - DH)];
                    fr[col] = xv; fr[66 + col] = 0.0f;
                    fr[132 + col] = -xv; fr[198 + col] = 0.0f; fr[264 + col] = -xv;
                }
            }
        }
    }
}

// ---------------------------------------------------------------------------
// cand GEMM: c = tanh(feat @ Wc + bc); h = u*h + (1-u)*c; out[:,t] = h
// ---------------------------------------------------------------------------
__global__ void __launch_bounds__(256)
dcgru_gemm_cand_kernel(const float* __restrict__ feat,
                       const _Float16* __restrict__ Wf,
                       const float* __restrict__ bc,
                       float* __restrict__ h,
                       const float* __restrict__ u,
                       float* __restrict__ out,
                       int t) {
    __shared__ __align__(32) _Float16 sA[128 * 32];
    v8f acc[NCAND / 16];
    const int rowbase = blockIdx.x * 128;

    gemm_block<NCAND>(feat, Wf, rowbase, acc, sA);

    const int lane = threadIdx.x & 31;
    const int wave = threadIdx.x >> 5;
    const int lh   = lane >> 4;
    const int l15  = lane & 15;
    const int R0   = rowbase + wave * 16;

    #pragma unroll
    for (int nt = 0; nt < NCAND / 16; ++nt) {
        int col = nt * 16 + l15;
        float bias = bc[col];
        #pragma unroll
        for (int r = 0; r < 8; ++r) {
            int R = R0 + r + 8 * lh;
            float cval = tanhf(acc[nt][r] + bias);
            float uv = u[(size_t)R * DH + col];
            float hp = h[(size_t)R * DH + col];
            float hn = uv * hp + (1.0f - uv) * cval;
            h[(size_t)R * DH + col] = hn;
            int b = R / DN, node = R - b * DN;
            out[(((size_t)b * DL + t) * DN + node) * DH + col] = hn;
        }
    }
}

// final hidden state -> second output slot
__global__ void dcgru_copy_hn_kernel(float* __restrict__ dst,
                                     const float* __restrict__ h) {
    int idx = blockIdx.x * blockDim.x + threadIdx.x;
    if (idx < DM * DH) dst[idx] = h[idx];
}

// ---------------------------------------------------------------------------
extern "C" void kernel_launch(void* const* d_in, const int* in_sizes, int n_in,
                              void* d_out, int out_size, void* d_ws, size_t ws_size,
                              hipStream_t stream) {
    const float* x      = (const float*)d_in[0];
    const int*   As_idx = (const int*)  d_in[1];
    const float* As_w   = (const float*)d_in[2];
    const float* h0     = (const float*)d_in[3];
    const float* Wg     = (const float*)d_in[4];
    const float* bg     = (const float*)d_in[5];
    const float* Wc     = (const float*)d_in[6];
    const float* bc     = (const float*)d_in[7];
    float* out = (float*)d_out;
    float* hn_out = out + (size_t)DB * DL * DN * DH;

    float*    feat = (float*)d_ws;                      // 64000 x 352 fp32
    float*    h    = feat + (size_t)DM * DKP;           // 64000 x 64
    float*    u    = h    + (size_t)DM * DH;            // 64000 x 64
    _Float16* WfG  = (_Float16*)(u + (size_t)DM * DH);  // 11*8*32*16 f16
    _Float16* WfC  = WfG + DKC * (NGATE / 16) * 512;    // 11*4*32*16 f16

    const int blk = 256;
    const int nWfG = DKC * (NGATE / 16) * 512;  // 45056
    const int nWfC = DKC * (NCAND / 16) * 512;  // 22528

    dcgru_init_kernel<<<(DM * DH + DM * (DKP - DKD) + blk - 1) / blk, blk, 0, stream>>>(h, h0, feat);
    dcgru_pack_w_kernel<NGATE><<<nWfG / blk, blk, 0, stream>>>(Wg, WfG);
    dcgru_pack_w_kernel<NCAND><<<nWfC / blk, blk, 0, stream>>>(Wc, WfC);

    dim3 spmm_grid((DE * DCAT + blk - 1) / blk, DS);
    for (int t = 0; t < DL; ++t) {
        dcgru_concat_kernel<<<(DM * DCAT + blk - 1) / blk, blk, 0, stream>>>(feat, h, x, t);
        dcgru_spmm_kernel<<<spmm_grid, blk, 0, stream>>>(feat, As_idx, As_w, 1);
        dcgru_spmm_kernel<<<spmm_grid, blk, 0, stream>>>(feat, As_idx, As_w, 2);
        dcgru_gemm_gate_kernel<<<DM / 128, blk, 0, stream>>>(feat, WfG, bg, h, x, u, t);
        dcgru_spmm_kernel<<<spmm_grid, blk, 0, stream>>>(feat, As_idx, As_w, 1);
        dcgru_spmm_kernel<<<spmm_grid, blk, 0, stream>>>(feat, As_idx, As_w, 2);
        dcgru_gemm_cand_kernel<<<DM / 128, blk, 0, stream>>>(feat, WfC, bc, h, u, out, t);
    }
    dcgru_copy_hn_kernel<<<(DM * DH + blk - 1) / blk, blk, 0, stream>>>(hn_out, h);
}
